// MultiHeadAttention_11227044512077
// MI455X (gfx1250) — matrix operand order repaired and checked by reading, not compiled, not run on hardware
//
#include <hip/hip_runtime.h>

typedef __attribute__((ext_vector_type(16))) _Float16 v16h;
typedef __attribute__((ext_vector_type(8)))  _Float16 v8h;
typedef __attribute__((ext_vector_type(8)))  float    v8f;
typedef int v4i_async __attribute__((vector_size(16)));

#define HIDDEN 512
#define HEADS  8
#define HD     64
#define BATCH  2
#define SEQ    4096
#define MTOT   (BATCH * SEQ)   // 8192 rows

// ---------------------------------------------------------------------------
// gfx1250 async global->LDS copy path (ASYNCcnt-tracked), with compile-safe
// fallback to load + ds_store staging if the builtins are not declared.
// Probe-confirmed signature: (as1 v4i*, as3 v4i*, imm offset, imm cpol).
// ---------------------------------------------------------------------------
#if defined(__gfx1250__) && __has_builtin(__builtin_amdgcn_global_load_async_to_lds_b128)
#define HAS_ASYNC_LDS 1
#else
#define HAS_ASYNC_LDS 0
#endif

#if HAS_ASYNC_LDS
#define ASYNC_COPY_B128(gsrc, ldst)                                           \
  __builtin_amdgcn_global_load_async_to_lds_b128(                             \
      (__attribute__((address_space(1))) v4i_async*)(gsrc),                   \
      (__attribute__((address_space(3))) v4i_async*)(ldst), 0, 0)
#if __has_builtin(__builtin_amdgcn_s_wait_asynccnt)
#define WAIT_ASYNC(n) __builtin_amdgcn_s_wait_asynccnt(n)
#else
#define WAIT_ASYNC(n) asm volatile("s_wait_asynccnt %0" ::"n"(n) : "memory")
#endif
#else
#define ASYNC_COPY_B128(gsrc, ldst) (*(v8h*)(ldst) = *(const v8h*)(gsrc))
#define WAIT_ASYNC(n)
#endif

// ---------------------------------------------------------------------------
// Prep: f32 -> f16 convert, and weight transpose (W[k][n] -> Wt[n][k]) so the
// WMMA B-fragment (lane = output column, K contiguous) is a contiguous load.
// ---------------------------------------------------------------------------
__global__ void cvt_f32_to_f16(const float* __restrict__ src,
                               _Float16* __restrict__ dst, int n) {
  int i = blockIdx.x * blockDim.x + threadIdx.x;
  if (i < n) dst[i] = (_Float16)src[i];
}

__global__ void transpose_w(const float* __restrict__ w,
                            _Float16* __restrict__ wt) {
  // 512x512, block = 256 threads = 16x16 tile
  int k = blockIdx.x * 16 + (threadIdx.x & 15);
  int n = blockIdx.y * 16 + (threadIdx.x >> 4);
  wt[(size_t)n * HIDDEN + k] = (_Float16)w[(size_t)k * HIDDEN + n];
}

// ---------------------------------------------------------------------------
// WMMA GEMM: out[r][c] = sum_k A[r][k] * Bt[c][k] + bias[c]
// A: [M,512] f16 row-major.  Bt: [512,512] f16, Bt[n][k].
// 8 waves/block, wave = 16 rows x 64 cols (4 C-tiles). Grid (N/64, M/128).
// Weight tiles staged in LDS via double-buffered async copies (shared by all
// 8 waves). All 4 B fragments batch-loaded before the WMMA sequence so
// ds_load latency overlaps WMMA issue instead of s_wait_dscnt 0 per WMMA.
// mode 0: f16 out [M,512]   (Q, K)
// mode 1: f32 out [M,512]   (final projection -> d_out)
// mode 2: f16 out transposed per head: Vt[b][h][d][s]   (V)
// ---------------------------------------------------------------------------
__global__ __launch_bounds__(256) void gemm_wmma(
    const _Float16* __restrict__ A, const _Float16* __restrict__ Bt,
    const float* __restrict__ bias, void* __restrict__ out, int mode) {
  __shared__ alignas(16) _Float16 bsh[2][64][32];  // [buf][n in 64-col blk][k]
  const int tid    = threadIdx.x;
  const int lane   = tid & 31;
  const int wave   = tid >> 5;
  const int half16 = lane >> 4;
  const int l16    = lane & 15;
  const int rbase  = blockIdx.y * 128 + wave * 16;
  const int cbase  = blockIdx.x * 64;

  // staging map: thread t -> 16B of the 64x32 f16 weight tile
  const int sn = tid >> 2;          // 0..63
  const int sk = (tid & 3) * 8;     // 0,8,16,24
  const _Float16* Bstage = Bt + (size_t)(cbase + sn) * HIDDEN + sk;

  v8f acc[4] = {};
  const _Float16* Arow = A + (size_t)(rbase + l16) * HIDDEN;

  ASYNC_COPY_B128(Bstage, &bsh[0][sn][sk]);

  for (int k0 = 0; k0 < HIDDEN; k0 += 32) {
    const int cur = (k0 >> 5) & 1;
    if (k0 + 32 < HIDDEN) {
      ASYNC_COPY_B128(Bstage + k0 + 32, &bsh[cur ^ 1][sn][sk]);
      WAIT_ASYNC(1);
    } else {
      WAIT_ASYNC(0);
    }
    __syncthreads();

    // A fragment (wave-private, direct global): lane row = l16,
    // K chunks [k0+8h, +8) and [k0+16+8h, +8)
    union { v16h v; v8h p[2]; } a;
    a.p[0] = *(const v8h*)(Arow + k0 + half16 * 8);
    a.p[1] = *(const v8h*)(Arow + k0 + 16 + half16 * 8);

    // Batch-load all 4 B fragments from LDS (distinct regs), then 4 WMMAs.
    union { v16h v; v8h p[2]; } bf[4];
#pragma unroll
    for (int nt = 0; nt < 4; ++nt) {
      bf[nt].p[0] = *(const v8h*)(&bsh[cur][nt * 16 + l16][half16 * 8]);
      bf[nt].p[1] = *(const v8h*)(&bsh[cur][nt * 16 + l16][16 + half16 * 8]);
    }
#pragma unroll
    for (int nt = 0; nt < 4; ++nt)
      acc[nt] = __builtin_amdgcn_wmma_f32_16x16x32_f16(
          false, a.v, false, bf[nt].v, (short)0, acc[nt], false, false);
    __syncthreads();
  }

#pragma unroll
  for (int nt = 0; nt < 4; ++nt) {
    const int c = cbase + nt * 16 + l16;
    const float bb = bias[c];
#pragma unroll
    for (int j = 0; j < 8; ++j) {
      const int r = rbase + half16 * 8 + j;   // C layout: VGPR j -> row 8h+j
      const float val = acc[nt][j] + bb;
      if (mode == 0) {
        ((_Float16*)out)[(size_t)r * HIDDEN + c] = (_Float16)val;
      } else if (mode == 1) {
        ((float*)out)[(size_t)r * HIDDEN + c] = val;
      } else {
        const int b = r >> 12, s = r & (SEQ - 1);
        const int h = c >> 6,  d = c & 63;
        ((_Float16*)out)[((((size_t)b * HEADS + h) * HD + d) * SEQ) + s] =
            (_Float16)val;
      }
    }
  }
}

// ---------------------------------------------------------------------------
// Flash attention. Grid (S/128, H, B), 8 waves/block, wave = 16 query rows.
// Scores never touch memory. K/V tiles staged in LDS via double-buffered
// async copies (shared across all 8 waves). Online softmax with 16-lane
// shfl_xor row reductions; P transposed to A-layout via 1KB/wave LDS patch.
// K and V fragments batch-loaded ahead of their WMMA sequences.
// ---------------------------------------------------------------------------
__global__ __launch_bounds__(256) void flash_attn(
    const _Float16* __restrict__ Qh, const _Float16* __restrict__ Kh,
    const _Float16* __restrict__ Vt, _Float16* __restrict__ Oh) {
  __shared__ alignas(16) _Float16 ksh[2][32][64];  // K rows [kb..kb+32) x d
  __shared__ alignas(16) _Float16 vsh[2][64][32];  // V^T: d x s-block
  __shared__ alignas(16) _Float16 psh[8][16][32];  // per-wave P patch
  const int tid    = threadIdx.x;
  const int lane   = tid & 31;
  const int wave   = tid >> 5;
  const int half16 = lane >> 4;
  const int l16    = lane & 15;
  const int h    = blockIdx.y;
  const int b    = blockIdx.z;
  const int qrow = blockIdx.x * 128 + wave * 16;

  const _Float16* Kbase = Kh + (size_t)b * SEQ * HIDDEN + h * HD;
  const _Float16* Vbase = Vt + ((size_t)b * HEADS + h) * HD * SEQ;

  // staging maps (256 threads x 16B = 4KB per tile)
  const int kr = tid >> 3, kc = (tid & 7) * 8;   // K: row 0..31, col 0..56
  const int vd = tid >> 2, vs = (tid & 3) * 8;   // V: d 0..63, s-off 0..24
  const _Float16* Kstage = Kbase + (size_t)kr * HIDDEN + kc;
  const _Float16* Vstage = Vbase + (size_t)vd * SEQ + vs;

  // Q A-fragments for K-dim chunks [0,32) and [32,64) (wave-private)
  const size_t rowQ = (size_t)(b * SEQ + qrow + l16) * HIDDEN + h * HD;
  union { v16h v; v8h p[2]; } aq0, aq1;
  aq0.p[0] = *(const v8h*)(Qh + rowQ + half16 * 8);
  aq0.p[1] = *(const v8h*)(Qh + rowQ + 16 + half16 * 8);
  aq1.p[0] = *(const v8h*)(Qh + rowQ + 32 + half16 * 8);
  aq1.p[1] = *(const v8h*)(Qh + rowQ + 48 + half16 * 8);

  v8f o[4] = {};
  float m[8], ssum[8];
#pragma unroll
  for (int j = 0; j < 8; ++j) { m[j] = -1e30f; ssum[j] = 0.0f; }
  const float scale = 0.125f;  // 1/sqrt(64)

  ASYNC_COPY_B128(Kstage, &ksh[0][kr][kc]);
  ASYNC_COPY_B128(Vstage, &vsh[0][vd][vs]);

  for (int kb = 0; kb < SEQ; kb += 32) {
    const int cur = (kb >> 5) & 1;
    if (kb + 32 < SEQ) {
      ASYNC_COPY_B128(Kstage + (size_t)(kb + 32) * HIDDEN, &ksh[cur ^ 1][kr][kc]);
      ASYNC_COPY_B128(Vstage + (kb + 32), &vsh[cur ^ 1][vd][vs]);
      WAIT_ASYNC(2);
    } else {
      WAIT_ASYNC(0);
    }
    __syncthreads();

    // ---- two 16x16 score tiles: S = Q(16x64) @ K^T, K from LDS ----
    // Batch-load all 4 K fragments (2 tiles x 2 K-chunks), then 4 WMMAs.
    union { v16h v; v8h p[2]; } bk[2][2];
#pragma unroll
    for (int t = 0; t < 2; ++t) {
#pragma unroll
      for (int kk = 0; kk < 2; ++kk) {
        bk[t][kk].p[0] =
            *(const v8h*)(&ksh[cur][t * 16 + l16][kk * 32 + half16 * 8]);
        bk[t][kk].p[1] =
            *(const v8h*)(&ksh[cur][t * 16 + l16][kk * 32 + 16 + half16 * 8]);
      }
    }
    v8f sc[2];
#pragma unroll
    for (int t = 0; t < 2; ++t) {
      v8f c = {};
      c = __builtin_amdgcn_wmma_f32_16x16x32_f16(false, aq0.v, false,
                                                 bk[t][0].v, (short)0, c,
                                                 false, false);
      c = __builtin_amdgcn_wmma_f32_16x16x32_f16(false, aq1.v, false,
                                                 bk[t][1].v, (short)0, c,
                                                 false, false);
      sc[t] = c;
    }

    // ---- online softmax (row = 8*half16 + j, spread over 16 lanes) ----
    float corr[8];
#pragma unroll
    for (int j = 0; j < 8; ++j) {
      float s0 = sc[0][j] * scale, s1 = sc[1][j] * scale;
      float mx = fmaxf(s0, s1);
#pragma unroll
      for (int off = 1; off < 16; off <<= 1)
        mx = fmaxf(mx, __shfl_xor(mx, off, 32));
      const float nm = fmaxf(m[j], mx);
      corr[j] = __expf(m[j] - nm);
      m[j] = nm;
      const float p0 = __expf(s0 - nm);
      const float p1 = __expf(s1 - nm);
      float rs = p0 + p1;
#pragma unroll
      for (int off = 1; off < 16; off <<= 1)
        rs += __shfl_xor(rs, off, 32);
      ssum[j] = ssum[j] * corr[j] + rs;
      sc[0][j] = p0;
      sc[1][j] = p1;
    }
#pragma unroll
    for (int j = 0; j < 8; ++j) {
      o[0][j] *= corr[j]; o[1][j] *= corr[j];
      o[2][j] *= corr[j]; o[3][j] *= corr[j];
    }

    // ---- transpose P (C layout) -> A layout via per-wave LDS patch ----
#pragma unroll
    for (int j = 0; j < 8; ++j) {
      psh[wave][half16 * 8 + j][l16]      = (_Float16)sc[0][j];
      psh[wave][half16 * 8 + j][16 + l16] = (_Float16)sc[1][j];
    }
    __builtin_amdgcn_wave_barrier();
    union { v16h v; v8h p[2]; } pa;
    pa.p[0] = *(const v8h*)(&psh[wave][l16][half16 * 8]);
    pa.p[1] = *(const v8h*)(&psh[wave][l16][16 + half16 * 8]);

    // ---- O(16x64) += P(16x32) @ V(32x64), V^T tile from LDS ----
    // Batch-load all 4 V fragments, then 4 WMMAs.
    union { v16h v; v8h p[2]; } bv[4];
#pragma unroll
    for (int t = 0; t < 4; ++t) {
      bv[t].p[0] = *(const v8h*)(&vsh[cur][t * 16 + l16][half16 * 8]);
      bv[t].p[1] = *(const v8h*)(&vsh[cur][t * 16 + l16][16 + half16 * 8]);
    }
#pragma unroll
    for (int t = 0; t < 4; ++t)
      o[t] = __builtin_amdgcn_wmma_f32_16x16x32_f16(false, pa.v, false,
                                                    bv[t].v, (short)0, o[t],
                                                    false, false);
    __syncthreads();
  }

  // ---- normalize and write O as f16 [B*S, 512] ----
#pragma unroll
  for (int j = 0; j < 8; ++j) {
    const float inv = 1.0f / ssum[j];
    const int s = qrow + half16 * 8 + j;
    const size_t rowO = (size_t)(b * SEQ + s) * HIDDEN + h * HD;
#pragma unroll
    for (int t = 0; t < 4; ++t)
      Oh[rowO + t * 16 + l16] = (_Float16)(o[t][j] * inv);
  }
}

// ---------------------------------------------------------------------------
extern "C" void kernel_launch(void* const* d_in, const int* in_sizes, int n_in,
                              void* d_out, int out_size, void* d_ws,
                              size_t ws_size, hipStream_t stream) {
  const float* x  = (const float*)d_in[0];
  const float* Wq = (const float*)d_in[1];
  const float* bq = (const float*)d_in[2];
  const float* Wk = (const float*)d_in[3];
  const float* bk = (const float*)d_in[4];
  const float* Wv = (const float*)d_in[5];
  const float* bv = (const float*)d_in[6];
  const float* Wo = (const float*)d_in[7];
  const float* bo = (const float*)d_in[8];

  // workspace layout (f16), ~44 MB total
  _Float16* xh  = (_Float16*)d_ws;
  _Float16* Wqt = xh  + (size_t)MTOT * HIDDEN;
  _Float16* Wkt = Wqt + (size_t)HIDDEN * HIDDEN;
  _Float16* Wvt = Wkt + (size_t)HIDDEN * HIDDEN;
  _Float16* Wot = Wvt + (size_t)HIDDEN * HIDDEN;
  _Float16* Qh  = Wot + (size_t)HIDDEN * HIDDEN;
  _Float16* Kh  = Qh  + (size_t)MTOT * HIDDEN;
  _Float16* Vt  = Kh  + (size_t)MTOT * HIDDEN;
  _Float16* Oh  = Vt  + (size_t)MTOT * HIDDEN;

  const int n = MTOT * HIDDEN;
  cvt_f32_to_f16<<<(n + 255) / 256, 256, 0, stream>>>(x, xh, n);
  dim3 tg(32, 32);
  transpose_w<<<tg, 256, 0, stream>>>(Wq, Wqt);
  transpose_w<<<tg, 256, 0, stream>>>(Wk, Wkt);
  transpose_w<<<tg, 256, 0, stream>>>(Wv, Wvt);
  transpose_w<<<tg, 256, 0, stream>>>(Wo, Wot);

  dim3 gg(HIDDEN / 64, MTOT / 128);
  gemm_wmma<<<gg, 256, 0, stream>>>(xh, Wqt, bq, Qh, 0);
  gemm_wmma<<<gg, 256, 0, stream>>>(xh, Wkt, bk, Kh, 0);
  gemm_wmma<<<gg, 256, 0, stream>>>(xh, Wvt, bv, Vt, 2);

  dim3 ga(SEQ / 128, HEADS, BATCH);
  flash_attn<<<ga, 256, 0, stream>>>(Qh, Kh, Vt, Oh);

  gemm_wmma<<<gg, 256, 0, stream>>>(Oh, Wot, bo, (float*)d_out, 1);
}
